// EncoderProcessorDecoder_21036749816046
// MI455X (gfx1250) — matrix operand order repaired and checked by reading, not compiled, not run on hardware
//
#include <hip/hip_runtime.h>
#include <hip/hip_bf16.h>

// ---------------------------------------------------------------------------
// CDNA5 (gfx1250) fused GNN encode-process-decode.
// All GEMMs via v_wmma_f32_16x16x32_bf16 (fp32 accum). 3-layer MLPs fused in
// one kernel: packed weights staged in LDS (up to 160KB), inter-layer h kept
// in a per-wave padded LDS tile. Gather fused into layer-1 A loads; scatter
// fused into edge-MLP epilogue via global_atomic_add_f32.
// ---------------------------------------------------------------------------

typedef __attribute__((ext_vector_type(16))) __bf16 v16bf;
typedef __attribute__((ext_vector_type(8)))  float  v8f;

union AF { v16bf v; uint4 q[2]; };

__device__ __forceinline__ v8f wmma_bf16(const AF& a, const AF& b, v8f c) {
  return __builtin_amdgcn_wmma_f32_16x16x32_bf16(
      /*neg_a=*/false, a.v, /*neg_b=*/false, b.v,
      /*c_mod=*/(short)0, c, /*reuse_a=*/false, /*reuse_b=*/false);
}

// relu as ternary: folds to a single v_max_num_f32 (fmaxf emits an extra
// NaN-canonicalizing self-max).
__device__ __forceinline__ float relu(float x) { return x > 0.f ? x : 0.f; }

__device__ __forceinline__ void cvt4(AF& a, int base, float4 f) {
  a.v[base + 0] = (__bf16)f.x; a.v[base + 1] = (__bf16)f.y;
  a.v[base + 2] = (__bf16)f.z; a.v[base + 3] = (__bf16)f.w;
}

__device__ __forceinline__ void init_bias(v8f acc[8], const float* __restrict__ b, int r) {
#pragma unroll
  for (int j = 0; j < 8; ++j) {
    float bv = b[16 * j + r];
    v8f t;
#pragma unroll
    for (int i = 0; i < 8; ++i) t[i] = bv;
    acc[j] = t;
  }
}

// Write relu(acc) as bf16 into the per-wave row tile (stride 136 to dodge
// LDS bank conflicts; rows = C layout: VGPR v -> row half*8+v, col 16j+r).
__device__ __forceinline__ void store_relu(__bf16* srow, v8f acc[8], int half, int r) {
#pragma unroll
  for (int j = 0; j < 8; ++j) {
#pragma unroll
    for (int v = 0; v < 8; ++v) {
      srow[(half * 8 + v) * 136 + 16 * j + r] = (__bf16)relu(acc[j][v]);
    }
  }
}

// One 128-K GEMM layer whose A comes from the per-wave LDS row tile and whose
// B fragments come from packed weights in LDS.
__device__ __forceinline__ void gemm_lds(const __bf16* __restrict__ wlds,
                                         const __bf16* __restrict__ srow,
                                         v8f acc[8], int half, int r, int lane) {
#pragma unroll
  for (int kb = 0; kb < 4; ++kb) {
    AF a;
    const int base0 = kb * 32 + half * 8;  // A 16x32 bf16 layout per ISA table
    a.q[0] = *(const uint4*)(srow + r * 136 + base0);
    a.q[1] = *(const uint4*)(srow + r * 136 + base0 + 16);
#pragma unroll
    for (int j = 0; j < 8; ++j) {
      AF b;
      const __bf16* wp = wlds + ((kb * 8 + j) * 32 + lane) * 16;
      b.q[0] = *(const uint4*)wp;
      b.q[1] = *(const uint4*)(wp + 8);
      acc[j] = wmma_bf16(a, b, acc[j]);
    }
  }
}

// MODE: 0 = encoder (padded bf16 in, LN out)
//       1 = edge step (gather3 in, LN out + fp32 atomic scatter to agg)
//       2 = node step (agg fp32 + X bf16 in, LN out)
//       3 = decoder first 2 layers (bf16 in, relu out, no layer 3 / LN)
template <int K1P, int MODE>
__global__ __launch_bounds__(512)
void fused_mlp(const __bf16* __restrict__ in0,
               const __bf16* __restrict__ xmat,
               const float*  __restrict__ aggin,
               const int*    __restrict__ srcIdx,
               const int*    __restrict__ dstIdx,
               int M,
               const __bf16* __restrict__ pk1,
               const __bf16* __restrict__ pk2,
               const __bf16* __restrict__ pk3,
               const float* __restrict__ b1,
               const float* __restrict__ b2,
               const float* __restrict__ b3,
               const float* __restrict__ lng,
               const float* __restrict__ lnb,
               __bf16* __restrict__ out,
               float*  __restrict__ agg_out) {
  constexpr bool HAS_L3 = (MODE != 3);
  constexpr int W1E = K1P * 128;
  constexpr int W2E = 128 * 128;
  constexpr int W3E = HAS_L3 ? 128 * 128 : 0;

  extern __shared__ char smem[];
  __bf16* w1 = (__bf16*)smem;
  __bf16* w2 = w1 + W1E;
  __bf16* w3 = w2 + W2E;
  __bf16* shrow = w3 + W3E;  // 16 waves * 16 rows * 136 bf16

  const int lane = threadIdx.x & 31;
  const int wave = threadIdx.x >> 5;
  const int half = lane >> 4;
  const int r    = lane & 15;
  __bf16* srow = shrow + wave * (16 * 136);

  // --- stage packed weights into LDS (uint4 cooperative copy) ---
  {
    const uint4* s1 = (const uint4*)pk1; uint4* d1 = (uint4*)w1;
    for (int i = threadIdx.x; i < W1E / 8; i += (int)blockDim.x) d1[i] = s1[i];
    const uint4* s2 = (const uint4*)pk2; uint4* d2 = (uint4*)w2;
    for (int i = threadIdx.x; i < W2E / 8; i += (int)blockDim.x) d2[i] = s2[i];
    if constexpr (HAS_L3) {
      const uint4* s3 = (const uint4*)pk3; uint4* d3 = (uint4*)w3;
      for (int i = threadIdx.x; i < W3E / 8; i += (int)blockDim.x) d3[i] = s3[i];
    }
  }
  __syncthreads();

  const long rowBase = (long)blockIdx.x * 256;     // 16 waves * 16 rows
  const long rowA  = rowBase + wave * 16 + r;      // this lane's A-matrix row
  const long rowAL = (rowA < (long)M) ? rowA : (long)M - 1;

  int di = 0, si = 0;
  if constexpr (MODE == 1) { di = dstIdx[rowAL]; si = srcIdx[rowAL]; }

  v8f acc[8];
  init_bias(acc, b1, r);

  // ---------------- Layer 1 (K = K1P, gather fused) ----------------
#pragma unroll
  for (int kb = 0; kb < K1P / 32; ++kb) {
    AF a;
    if constexpr (MODE == 0) {
      const __bf16* p = in0 + rowAL * K1P + half * 8;  // K1P == 32 padded
      a.q[0] = *(const uint4*)p; a.q[1] = *(const uint4*)(p + 16);
    } else if constexpr (MODE == 1) {
      const __bf16* p; int colb;
      if (kb < 4)      { p = in0  + rowAL * 128;     colb = kb * 32; }
      else if (kb < 8) { p = xmat + (long)di * 128;  colb = (kb - 4) * 32; }
      else             { p = xmat + (long)si * 128;  colb = (kb - 8) * 32; }
      p += colb + half * 8;
      a.q[0] = *(const uint4*)p; a.q[1] = *(const uint4*)(p + 16);
    } else if constexpr (MODE == 2) {
      if (kb < 4) {
        const float* p = aggin + rowAL * 128 + kb * 32 + half * 8;
        cvt4(a, 0,  *(const float4*)(p));
        cvt4(a, 4,  *(const float4*)(p + 4));
        cvt4(a, 8,  *(const float4*)(p + 16));
        cvt4(a, 12, *(const float4*)(p + 20));
      } else {
        const __bf16* p = xmat + rowAL * 128 + (kb - 4) * 32 + half * 8;
        a.q[0] = *(const uint4*)p; a.q[1] = *(const uint4*)(p + 16);
      }
    } else {  // MODE == 3
      const __bf16* p = in0 + rowAL * 128 + kb * 32 + half * 8;
      a.q[0] = *(const uint4*)p; a.q[1] = *(const uint4*)(p + 16);
    }
#pragma unroll
    for (int j = 0; j < 8; ++j) {
      AF b;
      const __bf16* wp = w1 + ((kb * 8 + j) * 32 + lane) * 16;
      b.q[0] = *(const uint4*)wp;
      b.q[1] = *(const uint4*)(wp + 8);
      acc[j] = wmma_bf16(a, b, acc[j]);
    }
  }
  store_relu(srow, acc, half, r);
  __syncthreads();

  // ---------------- Layer 2 ----------------
  init_bias(acc, b2, r);
  gemm_lds(w2, srow, acc, half, r, lane);
  __syncthreads();

  if constexpr (HAS_L3) {
    store_relu(srow, acc, half, r);
    __syncthreads();

    // ---------------- Layer 3 ----------------
    init_bias(acc, b3, r);
    gemm_lds(w3, srow, acc, half, r, lane);
    __syncthreads();

    // ---------------- LayerNorm over 128 features per row ----------------
    float mu[8], rs[8];
    {
      float s[8], q[8];
#pragma unroll
      for (int v = 0; v < 8; ++v) {
        float t = 0.f, t2 = 0.f;
#pragma unroll
        for (int j = 0; j < 8; ++j) { float x = acc[j][v]; t += x; t2 += x * x; }
        s[v] = t; q[v] = t2;
      }
#pragma unroll
      for (int m = 1; m < 16; m <<= 1) {
#pragma unroll
        for (int v = 0; v < 8; ++v) {
          s[v] += __shfl_xor(s[v], m, 16);
          q[v] += __shfl_xor(q[v], m, 16);
        }
      }
#pragma unroll
      for (int v = 0; v < 8; ++v) {
        float m_  = s[v] * (1.f / 128.f);
        float var = q[v] * (1.f / 128.f) - m_ * m_;
        mu[v] = m_;
        rs[v] = rsqrtf(var + 1e-5f);
      }
    }

    int dstc[8];
    if constexpr (MODE == 1) {
#pragma unroll
      for (int v = 0; v < 8; ++v) {
        long grow = rowBase + wave * 16 + half * 8 + v;  // C-matrix row
        dstc[v] = (grow < (long)M) ? dstIdx[grow] : -1;
      }
    }

#pragma unroll
    for (int j = 0; j < 8; ++j) {
      float g  = lng[16 * j + r];
      float bb = lnb[16 * j + r];
#pragma unroll
      for (int v = 0; v < 8; ++v) {
        float y = (acc[j][v] - mu[v]) * rs[v] * g + bb;
        srow[(half * 8 + v) * 136 + 16 * j + r] = (__bf16)y;
        if constexpr (MODE == 1) {
          if (dstc[v] >= 0) {
            __hip_atomic_fetch_add(agg_out + (long)dstc[v] * 128 + 16 * j + r, y,
                                   __ATOMIC_RELAXED, __HIP_MEMORY_SCOPE_AGENT);
          }
        }
      }
    }
  } else {
    store_relu(srow, acc, half, r);
  }
  __syncthreads();

  // ---------------- coalesced bf16 store of cols 0..63 of the wave's 16x128 tile ----------
#pragma unroll
  for (int it = 0; it < 4; ++it) {
    int c = lane + 32 * it;       // 0..127: 16 rows x 8 uint4 chunks
    int rr = c >> 3, part = c & 7;
    long grow = rowBase + wave * 16 + rr;
    if (grow < (long)M) {
      uint4 t = *(const uint4*)(srow + rr * 136 + part * 8);
      *(uint4*)(out + grow * 128 + part * 8) = t;
    }
  }
}

// Pack fp32 W[K][128] into per-lane-contiguous WMMA B fragments (bf16).
// Element (kb, j, lane, e): k = 32kb + (lane<16 ? 0 : 8) + (e&7) + (e<8 ? 0 : 16), n = 16j + lane&15.
__global__ void k_pack(const float* __restrict__ W, __bf16* __restrict__ pk,
                       int Kvalid, int KP) {
  long i = (long)blockIdx.x * blockDim.x + threadIdx.x;
  long total = (long)KP * 128;
  if (i >= total) return;
  int e    = (int)(i & 15);
  int lane = (int)((i >> 4) & 31);
  int j    = (int)((i >> 9) & 7);
  int kb   = (int)(i >> 12);
  int k = kb * 32 + ((lane < 16) ? 0 : 8) + (e & 7) + ((e < 8) ? 0 : 16);
  int n = j * 16 + (lane & 15);
  float v = (k < Kvalid) ? W[(long)k * 128 + n] : 0.f;
  pk[i] = (__bf16)v;
}

// fp32 [M,Kin] -> bf16 [M,KP] zero-padded.
__global__ void k_padcast(const float* __restrict__ src, __bf16* __restrict__ dst,
                          int M, int Kin, int KP) {
  long i = (long)blockIdx.x * blockDim.x + threadIdx.x;
  if (i >= (long)M * KP) return;
  int  c  = (int)(i % KP);
  long rw = i / KP;
  dst[i] = (c < Kin) ? (__bf16)src[rw * Kin + c] : (__bf16)0.f;
}

__global__ void k_zero(float* __restrict__ p, long n) {
  long i = (long)blockIdx.x * blockDim.x + threadIdx.x;
  if (i < n) p[i] = 0.f;
}

// Final decoder layer: 128 -> 3, fp32 weights, bf16 activations.
__global__ void k_dec_final(const __bf16* __restrict__ h, const float* __restrict__ W,
                            const float* __restrict__ b, float* __restrict__ out, int N) {
  long i = (long)blockIdx.x * blockDim.x + threadIdx.x;
  if (i >= (long)N) return;
  const __bf16* hp = h + i * 128;
  float o0 = b[0], o1 = b[1], o2 = b[2];
#pragma unroll 8
  for (int k = 0; k < 128; ++k) {
    float x = (float)hp[k];
    o0 += x * W[k * 3 + 0];
    o1 += x * W[k * 3 + 1];
    o2 += x * W[k * 3 + 2];
  }
  out[i * 3 + 0] = o0; out[i * 3 + 1] = o1; out[i * 3 + 2] = o2;
}

// ---------------------------------------------------------------------------

extern "C" void kernel_launch(void* const* d_in, const int* in_sizes, int n_in,
                              void* d_out, int out_size, void* d_ws, size_t ws_size,
                              hipStream_t stream) {
  (void)n_in; (void)out_size; (void)ws_size;

  const float* x  = (const float*)d_in[0];
  const float* ea = (const float*)d_in[1];
  const int*   ei = (const int*)d_in[2];
  const int N = in_sizes[0] / 16;
  const int M = in_sizes[1] / 4;
  const int* srcIdx = ei;        // edge_index[0]
  const int* dstIdx = ei + M;    // edge_index[1]

  auto F = [&](int i) { return (const float*)d_in[i]; };
  // d_in layout (insertion-order flattening):
  // 0:x 1:edge_attr 2:edge_index
  // node_enc 3..10: W0 W1 W2 b0 b1 b2 ln_g ln_b
  // edge_enc 11..18
  // dec 19..24: W0 W1 W2 b0 b1 b2
  // proc step s: edge at 25+16s (W0 W1 W2 b0 b1 b2 g b), node at 25+16s+8

  char* base = (char*)d_ws;
  size_t off = 0;
  auto alloc = [&](size_t bytes) -> char* {
    off = (off + 255) & ~(size_t)255;
    char* r = base + off;
    off += bytes;
    return r;
  };

  auto pack = [&](const float* W, __bf16* pk, int Kvalid, int KP) {
    long total = (long)KP * 128;
    int g = (int)((total + 255) / 256);
    k_pack<<<g, 256, 0, stream>>>(W, pk, Kvalid, KP);
  };

  // ---- packed weight buffers ----
  __bf16* pkNE[3] = { (__bf16*)alloc(32 * 128 * 2),  (__bf16*)alloc(128 * 128 * 2), (__bf16*)alloc(128 * 128 * 2) };
  __bf16* pkEE[3] = { (__bf16*)alloc(32 * 128 * 2),  (__bf16*)alloc(128 * 128 * 2), (__bf16*)alloc(128 * 128 * 2) };
  __bf16* pkDC[2] = { (__bf16*)alloc(128 * 128 * 2), (__bf16*)alloc(128 * 128 * 2) };
  __bf16 *pkE1[5], *pkE2[5], *pkE3[5], *pkN1[5], *pkN2[5], *pkN3[5];
  for (int s = 0; s < 5; ++s) {
    pkE1[s] = (__bf16*)alloc(384 * 128 * 2);
    pkE2[s] = (__bf16*)alloc(128 * 128 * 2);
    pkE3[s] = (__bf16*)alloc(128 * 128 * 2);
    pkN1[s] = (__bf16*)alloc(256 * 128 * 2);
    pkN2[s] = (__bf16*)alloc(128 * 128 * 2);
    pkN3[s] = (__bf16*)alloc(128 * 128 * 2);
  }

  // ---- activation buffers ----
  __bf16* Xpad = (__bf16*)alloc((size_t)N * 32 * 2);
  __bf16* Epad = (__bf16*)alloc((size_t)M * 32 * 2);
  __bf16* Xb[2] = { (__bf16*)alloc((size_t)N * 128 * 2), (__bf16*)alloc((size_t)N * 128 * 2) };
  __bf16* Eb[2] = { (__bf16*)alloc((size_t)M * 128 * 2), (__bf16*)alloc((size_t)M * 128 * 2) };
  float*  agg   = (float*)alloc((size_t)N * 128 * 4);

  // ---- pack all weights (cheap: ~1.7 MB total) ----
  pack(F(3),  pkNE[0], 16, 32);   pack(F(4),  pkNE[1], 128, 128); pack(F(5),  pkNE[2], 128, 128);
  pack(F(11), pkEE[0], 4, 32);    pack(F(12), pkEE[1], 128, 128); pack(F(13), pkEE[2], 128, 128);
  pack(F(19), pkDC[0], 128, 128); pack(F(20), pkDC[1], 128, 128);
  for (int s = 0; s < 5; ++s) {
    int eb = 25 + 16 * s, nb = eb + 8;
    pack(F(eb),     pkE1[s], 384, 384);
    pack(F(eb + 1), pkE2[s], 128, 128);
    pack(F(eb + 2), pkE3[s], 128, 128);
    pack(F(nb),     pkN1[s], 256, 256);
    pack(F(nb + 1), pkN2[s], 128, 128);
    pack(F(nb + 2), pkN3[s], 128, 128);
  }

  // ---- pad/cast raw inputs ----
  {
    long t1 = (long)N * 32;
    k_padcast<<<(int)((t1 + 255) / 256), 256, 0, stream>>>(x, Xpad, N, 16, 32);
    long t2 = (long)M * 32;
    k_padcast<<<(int)((t2 + 255) / 256), 256, 0, stream>>>(ea, Epad, M, 4, 32);
  }

  const size_t SH_ROW = (size_t)16 * 16 * 136 * 2;  // per-block h tiles
  const size_t shENC  = ((size_t)32  * 128 + 2 * 128 * 128) * 2 + SH_ROW;
  const size_t shEDGE = ((size_t)384 * 128 + 2 * 128 * 128) * 2 + SH_ROW;
  const size_t shNODE = ((size_t)256 * 128 + 2 * 128 * 128) * 2 + SH_ROW;
  const size_t shDEC  = ((size_t)128 * 128 + 128 * 128) * 2 + SH_ROW;

  const int gN = (N + 255) / 256;
  const int gM = (M + 255) / 256;

  // ---- encoders ----
  fused_mlp<32, 0><<<gN, 512, shENC, stream>>>(
      Xpad, nullptr, nullptr, nullptr, nullptr, N,
      pkNE[0], pkNE[1], pkNE[2], F(6), F(7), F(8), F(9), F(10), Xb[0], nullptr);
  fused_mlp<32, 0><<<gM, 512, shENC, stream>>>(
      Epad, nullptr, nullptr, nullptr, nullptr, M,
      pkEE[0], pkEE[1], pkEE[2], F(14), F(15), F(16), F(17), F(18), Eb[0], nullptr);

  // ---- message-passing steps ----
  int xc = 0, ec = 0;
  for (int s = 0; s < 5; ++s) {
    int eb = 25 + 16 * s, nb = eb + 8;
    long na = (long)N * 128;
    k_zero<<<(int)((na + 255) / 256), 256, 0, stream>>>(agg, na);

    fused_mlp<384, 1><<<gM, 512, shEDGE, stream>>>(
        Eb[ec], Xb[xc], nullptr, srcIdx, dstIdx, M,
        pkE1[s], pkE2[s], pkE3[s],
        F(eb + 3), F(eb + 4), F(eb + 5), F(eb + 6), F(eb + 7),
        Eb[1 - ec], agg);
    ec ^= 1;

    fused_mlp<256, 2><<<gN, 512, shNODE, stream>>>(
        nullptr, Xb[xc], agg, nullptr, nullptr, N,
        pkN1[s], pkN2[s], pkN3[s],
        F(nb + 3), F(nb + 4), F(nb + 5), F(nb + 6), F(nb + 7),
        Xb[1 - xc], nullptr);
    xc ^= 1;
  }

  // ---- decoder: two 128x128 layers fused, then 128->3 epilogue ----
  __bf16* Hdec = Xb[1 - xc];
  fused_mlp<128, 3><<<gN, 512, shDEC, stream>>>(
      Xb[xc], nullptr, nullptr, nullptr, nullptr, N,
      pkDC[0], pkDC[1], nullptr, F(22), F(23), nullptr, nullptr, nullptr,
      Hdec, nullptr);
  k_dec_final<<<(N + 255) / 256, 256, 0, stream>>>(Hdec, F(21), F(24),
                                                   (float*)d_out, N);
}
